// Qwen3VLVisionTransformer_5463198400902
// MI455X (gfx1250) — matrix operand order repaired and checked by reading, not compiled, not run on hardware
//
#include <hip/hip_runtime.h>
#include <math.h>

// ---------------- problem constants (Qwen3-VL vision tower) ----------------
#define T_TOK 1024
#define D_HID 1152
#define NH    16
#define DH    72
#define DHP   128         // head_dim padded to 128: K and N exact for WMMA tiling
#define FF    4304
#define FFP   4352        // FF padded to multiple of 128
#define DEPTH 6
#define OUTD  3584
#define MERGE 4608
#define SEG   512         // two packed images of 512 tokens each

typedef __attribute__((ext_vector_type(16))) __bf16 v16bf;
typedef __attribute__((ext_vector_type(8)))  float  v8f;
typedef __attribute__((ext_vector_type(4)))  int    v4i;

struct __attribute__((aligned(16))) U4 { unsigned x, y, z, w; };

// optional gfx1250 async global->LDS DMA staging (ASYNCcnt-tracked)
#if defined(__has_builtin)
#  if __has_builtin(__builtin_amdgcn_global_load_async_to_lds_b128)
#    define USE_ASYNC_LDS 1
#  endif
#endif
#ifndef USE_ASYNC_LDS
#  define USE_ASYNC_LDS 0
#endif

#if USE_ASYNC_LDS
typedef __attribute__((address_space(1))) v4i* gv4i_p;   // global (device) v4i*
typedef __attribute__((address_space(3))) v4i* lv4i_p;   // LDS (shared) v4i*
#endif

__device__ __forceinline__ void stage16(__bf16* dstLds, const __bf16* srcG) {
#if USE_ASYNC_LDS
  __builtin_amdgcn_global_load_async_to_lds_b128(
      (gv4i_p)const_cast<__bf16*>(srcG),
      (lv4i_p)dstLds, 0, 0);
#else
  *(U4*)dstLds = *(const U4*)srcG;
#endif
}
__device__ __forceinline__ void stage_wait() {
#if USE_ASYNC_LDS
#  if __has_builtin(__builtin_amdgcn_s_wait_asynccnt)
  __builtin_amdgcn_s_wait_asynccnt(0);
#  else
  asm volatile("s_wait_asynccnt 0" ::: "memory");
#  endif
#endif
}

__device__ __forceinline__ float gelu_tanh(float x) {
  return 0.5f * x * (1.0f + tanhf(0.7978845608028654f * (x + 0.044715f * x * x * x)));
}

// ---------------- prep / elementwise kernels ----------------
__global__ void k_rotary(const float* __restrict__ rot,
                         float* __restrict__ cosb, float* __restrict__ sinb) {
  int idx = blockIdx.x * blockDim.x + threadIdx.x;
  if (idx >= T_TOK * DH) return;
  int t = idx / DH, j = idx % DH;
  float a = rot[t * (DH / 2) + (j % (DH / 2))];  // emb = concat(rot, rot)
  cosb[idx] = cosf(a);
  sinb[idx] = sinf(a);
}

// (T, c,t,p,q) f32 -> (T, (t,p,q,c)) bf16 to match w_patch contraction order
__global__ void k_repack_pixels(const float* __restrict__ px, __bf16* __restrict__ dst) {
  int idx = blockIdx.x * blockDim.x + threadIdx.x;
  if (idx >= T_TOK * 1536) return;
  int l = idx / 1536, kk = idx % 1536;
  int c = kk % 3, q = (kk / 3) % 16, p = (kk / 48) % 16, t = kk / 768;
  dst[idx] = (__bf16)px[l * 1536 + c * 512 + t * 256 + p * 16 + q];
}

// (K,N) f32 -> (Np,Kp) bf16 zero-padded transpose (B^T layout for WMMA)
__global__ void k_transpose_pad(const float* __restrict__ src, __bf16* __restrict__ dst,
                                int K, int N, int Kp, int Np) {
  long idx = (long)blockIdx.x * blockDim.x + threadIdx.x;
  if (idx >= (long)Kp * Np) return;
  int n = (int)(idx / Kp), k = (int)(idx % Kp);
  float v = (n < N && k < K) ? src[(long)k * N + n] : 0.0f;
  dst[idx] = (__bf16)v;
}

// w_o (1152,1152) -> (1152, NH*DHP) bf16, K padded per head 72->128 with zeros
__global__ void k_transpose_wo(const float* __restrict__ src, __bf16* __restrict__ dst) {
  int idx = blockIdx.x * blockDim.x + threadIdx.x;
  if (idx >= D_HID * (NH * DHP)) return;
  int n = idx / (NH * DHP), kp = idx % (NH * DHP);
  int h = kp / DHP, d = kp % DHP;
  float v = (d < DH) ? src[(long)(h * DH + d) * D_HID + n] : 0.0f;
  dst[idx] = (__bf16)v;
}

__global__ __launch_bounds__(256) void k_layernorm(const float* __restrict__ x,
    const float* __restrict__ s, const float* __restrict__ b,
    __bf16* __restrict__ out, int Drow) {
  __shared__ float red[256];
  int row = blockIdx.x, tid = threadIdx.x;
  const float* xr = x + (long)row * Drow;
  float sum = 0.f;
  for (int i = tid; i < Drow; i += 256) sum += xr[i];
  red[tid] = sum; __syncthreads();
  for (int st = 128; st > 0; st >>= 1) { if (tid < st) red[tid] += red[tid + st]; __syncthreads(); }
  float mu = red[0] / Drow;
  __syncthreads();
  float vs = 0.f;
  for (int i = tid; i < Drow; i += 256) { float d = xr[i] - mu; vs += d * d; }
  red[tid] = vs; __syncthreads();
  for (int st = 128; st > 0; st >>= 1) { if (tid < st) red[tid] += red[tid + st]; __syncthreads(); }
  float rs = rsqrtf(red[0] / Drow + 1e-6f);
  for (int i = tid; i < Drow; i += 256)
    out[(long)row * Drow + i] = (__bf16)((xr[i] - mu) * rs * s[i] + b[i]);
}

// qkv (T,3456) f32 -> rope'd q,k (H,T,128) bf16 (zero-padded) and V^T (H,128,T) bf16
__global__ void k_rope_pack(const float* __restrict__ qkv,
                            const float* __restrict__ cosb, const float* __restrict__ sinb,
                            __bf16* __restrict__ qp, __bf16* __restrict__ kp,
                            __bf16* __restrict__ vT) {
  int idx = blockIdx.x * blockDim.x + threadIdx.x;
  if (idx >= NH * T_TOK * DHP) return;
  int d = idx % DHP, t = (idx / DHP) % T_TOK, h = idx / (DHP * T_TOK);
  float qv = 0.f, kv = 0.f, vv = 0.f;
  if (d < DH) {
    const float* rowp = qkv + (long)t * (3 * D_HID);
    float q0 = rowp[h * DH + d];
    float k0 = rowp[D_HID + h * DH + d];
    vv       = rowp[2 * D_HID + h * DH + d];
    int d2 = (d < DH / 2) ? d + DH / 2 : d - DH / 2;      // rotate_half
    float qr = rowp[h * DH + d2];
    float kr = rowp[D_HID + h * DH + d2];
    if (d < DH / 2) { qr = -qr; kr = -kr; }
    float c = cosb[t * DH + d], sn = sinb[t * DH + d];
    qv = q0 * c + qr * sn;
    kv = k0 * c + kr * sn;
  }
  qp[idx] = (__bf16)qv;
  kp[idx] = (__bf16)kv;
  vT[(long)h * DHP * T_TOK + (long)d * T_TOK + t] = (__bf16)vv;
}

// softmax over 512-wide rows of S (scale folded in); output bf16 probs
__global__ __launch_bounds__(256) void k_softmax(const float* __restrict__ S,
                                                 __bf16* __restrict__ P) {
  __shared__ float red[256];
  long base = (long)blockIdx.x * SEG;
  int tid = threadIdx.x;
  const float scale = 0.11785113019775793f;  // 1/sqrt(72)
  float a0 = S[base + tid] * scale;
  float a1 = S[base + tid + 256] * scale;
  red[tid] = fmaxf(a0, a1); __syncthreads();
  for (int st = 128; st > 0; st >>= 1) { if (tid < st) red[tid] = fmaxf(red[tid], red[tid + st]); __syncthreads(); }
  float mx = red[0]; __syncthreads();
  float e0 = __expf(a0 - mx), e1 = __expf(a1 - mx);
  red[tid] = e0 + e1; __syncthreads();
  for (int st = 128; st > 0; st >>= 1) { if (tid < st) red[tid] += red[tid + st]; __syncthreads(); }
  float inv = 1.0f / red[0];
  P[base + tid]       = (__bf16)(e0 * inv);
  P[base + tid + 256] = (__bf16)(e1 * inv);
}

// ------- bf16 WMMA GEMM, guard-free: M%64==0, N%128==0, K%32==0 required -------
#define BM 64
#define BN 128
#define BK 32

template<bool BIAS, bool RES, bool GELU, bool SF32, bool SBF16>
__global__ __launch_bounds__(128) void k_gemm(
    const __bf16* __restrict__ A, int lda,      // (M,K) row-major
    const __bf16* __restrict__ Bt, int ldb,     // (N,K) row-major  (= B^T)
    float* __restrict__ C, __bf16* __restrict__ Cb, int ldc,
    const float* __restrict__ bias, int Nb,     // bias valid for n < Nb
    const float* __restrict__ res, int ldr,
    int M, int N, int K,
    long sA, long sA2, long sB, long sB2, long sC, long sC2, int zdiv)
{
  __shared__ __attribute__((aligned(16))) __bf16 As[BM][BK];
  __shared__ __attribute__((aligned(16))) __bf16 Bs[BN][BK];
  int z = blockIdx.z;
  int zh = z / zdiv, zs = z % zdiv;
  A  += (long)zh * sA + (long)zs * sA2;
  Bt += (long)zh * sB + (long)zs * sB2;
  long cOff = (long)zh * sC + (long)zs * sC2;

  int m0 = blockIdx.y * BM;
  int n0 = blockIdx.x * BN;
  int tid = threadIdx.x;
  int lane = tid & 31, wv = tid >> 5;
  int wm = (wv & 1) * 32, wn = (wv >> 1) * 64;   // wave's 32x64 subtile
  int fm = lane & 15, fs = lane >> 4;            // fragment lane decode

  v8f acc[2][4] = {};

  for (int k0 = 0; k0 < K; k0 += BK) {
    // stage A (64x32) and Bt (128x32) tiles: pure 16B transfers, no guards
    for (int c = tid; c < 256; c += 128) {
      int row = c >> 2, col8 = (c & 3) * 8;
      stage16(&As[row][col8], A + (long)(m0 + row) * lda + (k0 + col8));
    }
    for (int c = tid; c < 512; c += 128) {
      int row = c >> 2, col8 = (c & 3) * 8;
      stage16(&Bs[row][col8], Bt + (long)(n0 + row) * ldb + (k0 + col8));
    }
    if (k0 + BK < K) {   // global_prefetch_b8 for next K tile
      __builtin_prefetch(A + (long)(m0 + (tid & 63)) * lda + k0 + BK, 0, 0);
      __builtin_prefetch(Bt + (long)(n0 + tid) * ldb + k0 + BK, 0, 0);
    }
    stage_wait();
    __syncthreads();

    // fragments per ISA layouts: A 16x32 (m=lane&15, khalf=lane>>4),
    // B 32x16 (n=lane&15, khalf=lane>>4) -> two ds_load_b128 each
    v16bf af[2], bfv[4];
    for (int i = 0; i < 2; ++i) {
      int m = wm + i * 16 + fm;
      U4* pa = (U4*)&af[i];
      pa[0] = *(const U4*)&As[m][fs * 8];
      pa[1] = *(const U4*)&As[m][16 + fs * 8];
    }
    for (int j = 0; j < 4; ++j) {
      int n = wn + j * 16 + fm;
      U4* pb = (U4*)&bfv[j];
      pb[0] = *(const U4*)&Bs[n][fs * 8];
      pb[1] = *(const U4*)&Bs[n][16 + fs * 8];
    }
    for (int i = 0; i < 2; ++i)
      for (int j = 0; j < 4; ++j)
        acc[i][j] = __builtin_amdgcn_wmma_f32_16x16x32_bf16(
            false, af[i], false, bfv[j], (short)0, acc[i][j], false, false);
    __syncthreads();
  }

  // epilogue; C layout: n=lane&15, m=(lane>>4)*8 + r
  int cm = fs * 8, cn = fm;
  for (int i = 0; i < 2; ++i)
    for (int j = 0; j < 4; ++j) {
      int nidx = n0 + wn + j * 16 + cn;
      float bv = 0.f;
      if (BIAS) bv = (nidx < Nb) ? bias[nidx] : 0.f;
      for (int r = 0; r < 8; ++r) {
        int mi = m0 + wm + i * 16 + cm + r;
        float v = acc[i][j][r] + bv;
        if (RES)  v += res[(long)mi * ldr + nidx];
        if (GELU) v = gelu_tanh(v);
        long off = cOff + (long)mi * ldc + nidx;
        if (SF32)  C[off]  = v;
        if (SBF16) Cb[off] = (__bf16)v;
      }
    }
}

// ---------------- orchestration ----------------
extern "C" void kernel_launch(void* const* d_in, const int* in_sizes, int n_in,
                              void* d_out, int out_size, void* d_ws, size_t ws_size,
                              hipStream_t stream) {
  const float* pixel   = (const float*)d_in[0];
  const float* pos     = (const float*)d_in[1];
  const float* rot     = (const float*)d_in[2];
  /* d_in[3] cu_seqlens: fixed [0,512,1024]; segmentation hardcoded as 2x512 */
  const float* w_patch = (const float*)d_in[4];
  const float* b_patch = (const float*)d_in[5];
  const float* w_qkv   = (const float*)d_in[6];
  const float* b_qkv   = (const float*)d_in[7];
  const float* w_o     = (const float*)d_in[8];
  const float* b_o     = (const float*)d_in[9];
  const float* ln1_s   = (const float*)d_in[10];
  const float* ln1_b   = (const float*)d_in[11];
  const float* ln2_s   = (const float*)d_in[12];
  const float* ln2_b   = (const float*)d_in[13];
  const float* w_fc1   = (const float*)d_in[14];
  const float* b_fc1   = (const float*)d_in[15];
  const float* w_fc2   = (const float*)d_in[16];
  const float* b_fc2   = (const float*)d_in[17];
  const float* mgls    = (const float*)d_in[18];
  const float* mglb    = (const float*)d_in[19];
  const float* mg_w1   = (const float*)d_in[20];
  const float* mg_b1   = (const float*)d_in[21];
  const float* mg_w2   = (const float*)d_in[22];
  const float* mg_b2   = (const float*)d_in[23];
  float* out = (float*)d_out;
  (void)in_sizes; (void)n_in; (void)out_size; (void)ws_size;

  char* p = (char*)d_ws;
  auto take = [&](size_t bytes) { char* r = p; p += (bytes + 255) & ~(size_t)255; return r; };
  __bf16* WB   = (__bf16*)take((size_t)MERGE * MERGE * 2);            // weight scratch (max 42.5MB)
  __bf16* pixb = (__bf16*)take((size_t)T_TOK * 1536 * 2);
  float*  xbuf = (float*) take((size_t)T_TOK * D_HID * 4);            // residual stream f32
  __bf16* hb   = (__bf16*)take((size_t)T_TOK * D_HID * 2);            // LN output bf16
  float*  qkvb = (float*) take((size_t)T_TOK * 3 * D_HID * 4);
  __bf16* qp   = (__bf16*)take((size_t)NH * T_TOK * DHP * 2);
  __bf16* kp   = (__bf16*)take((size_t)NH * T_TOK * DHP * 2);
  __bf16* vT   = (__bf16*)take((size_t)NH * DHP * T_TOK * 2);
  float*  Sb   = (float*) take((size_t)NH * 2 * SEG * SEG * 4);
  __bf16* Pb   = (__bf16*)take((size_t)NH * 2 * SEG * SEG * 2);
  __bf16* obf  = (__bf16*)take((size_t)T_TOK * (NH * DHP) * 2);
  __bf16* ffb  = (__bf16*)take((size_t)T_TOK * FFP * 2);
  __bf16* g1b  = (__bf16*)take((size_t)(T_TOK / 4) * MERGE * 2);
  float*  cosb = (float*) take((size_t)T_TOK * DH * 4);
  float*  sinb = (float*) take((size_t)T_TOK * DH * 4);

  auto cdiv = [](int a, int b) { return (a + b - 1) / b; };

  k_rotary<<<cdiv(T_TOK * DH, 256), 256, 0, stream>>>(rot, cosb, sinb);
  k_repack_pixels<<<cdiv(T_TOK * 1536, 256), 256, 0, stream>>>(pixel, pixb);

  // patch embed: x = pix @ w_patch + b_patch + pos_embed
  k_transpose_pad<<<cdiv(1536 * D_HID, 256), 256, 0, stream>>>(w_patch, WB, 1536, D_HID, 1536, D_HID);
  k_gemm<true, true, false, true, false><<<dim3(D_HID / BN, T_TOK / BM, 1), 128, 0, stream>>>(
      pixb, 1536, WB, 1536, xbuf, nullptr, D_HID, b_patch, D_HID, pos, D_HID,
      T_TOK, D_HID, 1536, 0, 0, 0, 0, 0, 0, 1);

  for (int L = 0; L < DEPTH; ++L) {
    k_layernorm<<<T_TOK, 256, 0, stream>>>(xbuf, ln1_s + L * D_HID, ln1_b + L * D_HID, hb, D_HID);
    k_transpose_pad<<<cdiv(D_HID * 3 * D_HID, 256), 256, 0, stream>>>(
        w_qkv + (size_t)L * D_HID * 3 * D_HID, WB, D_HID, 3 * D_HID, D_HID, 3 * D_HID);
    k_gemm<true, false, false, true, false><<<dim3(3 * D_HID / BN, T_TOK / BM, 1), 128, 0, stream>>>(
        hb, D_HID, WB, D_HID, qkvb, nullptr, 3 * D_HID, b_qkv + (size_t)L * 3 * D_HID, 3 * D_HID,
        nullptr, 0, T_TOK, 3 * D_HID, D_HID, 0, 0, 0, 0, 0, 0, 1);
    k_rope_pack<<<cdiv(NH * T_TOK * DHP, 256), 256, 0, stream>>>(qkvb, cosb, sinb, qp, kp, vT);

    // scores[h,seg] = q_seg @ k_seg^T   (z = h*2+seg, 32 batches; K=128 padded)
    k_gemm<false, false, false, true, false><<<dim3(SEG / BN, SEG / BM, NH * 2), 128, 0, stream>>>(
        qp, DHP, kp, DHP, Sb, nullptr, SEG, nullptr, 0, nullptr, 0,
        SEG, SEG, DHP,
        (long)T_TOK * DHP, (long)SEG * DHP, (long)T_TOK * DHP, (long)SEG * DHP,
        (long)2 * SEG * SEG, (long)SEG * SEG, 2);
    k_softmax<<<NH * 2 * SEG, 256, 0, stream>>>(Sb, Pb);
    // o[h,seg] = P @ V  -> obf (T, 16*128) bf16 (padded head layout)
    k_gemm<false, false, false, false, true><<<dim3(DHP / BN, SEG / BM, NH * 2), 128, 0, stream>>>(
        Pb, SEG, vT, T_TOK, nullptr, obf, NH * DHP, nullptr, 0, nullptr, 0,
        SEG, DHP, SEG,
        (long)2 * SEG * SEG, (long)SEG * SEG, (long)DHP * T_TOK, (long)SEG,
        (long)DHP, (long)SEG * NH * DHP, 2);

    // o-proj (K padded 72->128 via zero-filled w_o^T) + residual
    k_transpose_wo<<<cdiv(D_HID * NH * DHP, 256), 256, 0, stream>>>(w_o + (size_t)L * D_HID * D_HID, WB);
    k_gemm<true, true, false, true, false><<<dim3(D_HID / BN, T_TOK / BM, 1), 128, 0, stream>>>(
        obf, NH * DHP, WB, NH * DHP, xbuf, nullptr, D_HID, b_o + (size_t)L * D_HID, D_HID,
        xbuf, D_HID, T_TOK, D_HID, NH * DHP, 0, 0, 0, 0, 0, 0, 1);

    // MLP (FF padded 4304->4352 with zero weights/bias; gelu(0)=0)
    k_layernorm<<<T_TOK, 256, 0, stream>>>(xbuf, ln2_s + L * D_HID, ln2_b + L * D_HID, hb, D_HID);
    k_transpose_pad<<<cdiv(D_HID * FFP, 256), 256, 0, stream>>>(
        w_fc1 + (size_t)L * D_HID * FF, WB, D_HID, FF, D_HID, FFP);
    k_gemm<true, false, true, false, true><<<dim3(FFP / BN, T_TOK / BM, 1), 128, 0, stream>>>(
        hb, D_HID, WB, D_HID, nullptr, ffb, FFP, b_fc1 + (size_t)L * FF, FF, nullptr, 0,
        T_TOK, FFP, D_HID, 0, 0, 0, 0, 0, 0, 1);
    k_transpose_pad<<<cdiv(FFP * D_HID, 256), 256, 0, stream>>>(
        w_fc2 + (size_t)L * FF * D_HID, WB, FF, D_HID, FFP, D_HID);
    k_gemm<true, true, false, true, false><<<dim3(D_HID / BN, T_TOK / BM, 1), 128, 0, stream>>>(
        ffb, FFP, WB, FFP, xbuf, nullptr, D_HID, b_fc2 + (size_t)L * D_HID, D_HID,
        xbuf, D_HID, T_TOK, D_HID, FFP, 0, 0, 0, 0, 0, 0, 1);
  }

  // merger: LN -> reshape (256,4608) -> gelu(h@W1+b1)@W2+b2
  k_layernorm<<<T_TOK, 256, 0, stream>>>(xbuf, mgls, mglb, hb, D_HID);
  k_transpose_pad<<<cdiv(MERGE * MERGE, 256), 256, 0, stream>>>(mg_w1, WB, MERGE, MERGE, MERGE, MERGE);
  k_gemm<true, false, true, false, true><<<dim3(MERGE / BN, (T_TOK / 4) / BM, 1), 128, 0, stream>>>(
      hb, MERGE, WB, MERGE, nullptr, g1b, MERGE, mg_b1, MERGE, nullptr, 0,
      T_TOK / 4, MERGE, MERGE, 0, 0, 0, 0, 0, 0, 1);
  k_transpose_pad<<<cdiv(MERGE * OUTD, 256), 256, 0, stream>>>(mg_w2, WB, MERGE, OUTD, MERGE, OUTD);
  k_gemm<true, false, false, true, false><<<dim3(OUTD / BN, (T_TOK / 4) / BM, 1), 128, 0, stream>>>(
      g1b, MERGE, WB, MERGE, out, nullptr, OUTD, mg_b2, OUTD, nullptr, 0,
      T_TOK / 4, OUTD, MERGE, 0, 0, 0, 0, 0, 0, 1);
}
